// QTensorLinear_63161789055350
// MI455X (gfx1250) — matrix operand order repaired and checked by reading, not compiled, not run on hardware
//
#include <hip/hip_runtime.h>
#include <stdint.h>

// ---- toolchain feature detection -------------------------------------------
#if __has_include(<hip/amd_detail/amd_gfx1250_TDM.h>)
#define QTL_TDM_6ARG 1            // amdgpu-toolchain (therock): 6-arg builtin
#else
#define QTL_TDM_6ARG 0            // ROCm 7.2: 5-arg builtin
#endif

#if __has_builtin(__builtin_amdgcn_tensor_load_to_lds)
#define QTL_TDM 1
#else
#define QTL_TDM 0
#endif

typedef _Float16 v16h __attribute__((ext_vector_type(16)));
typedef _Float16 h8   __attribute__((ext_vector_type(8)));
typedef _Float16 h4   __attribute__((ext_vector_type(4)));
typedef float    v8f  __attribute__((ext_vector_type(8)));
typedef unsigned int u32x4 __attribute__((ext_vector_type(4)));
typedef int      i32x4 __attribute__((ext_vector_type(4)));
typedef int      i32x8 __attribute__((ext_vector_type(8)));

#define BM 128
#define BN 256
#define BK 32
#define KPAD 40          // 80B LDS row stride (TDM pad reproduces this), spreads banks
#define THREADS 256      // 8 waves (wave32): 2 (M) x 4 (N), each wave 64x64

// ============================================================================
// Kernel 1: one-time dequant/convert pass.
//   x  fp32 -> f16                  (ws region X, M*K*2 B)
//   qw int8-in-int32 -> f16 * sw    (ws region W, N*K*2 B)
// Both f16 operands (96 MB) fit the 192 MB L2, so the GEMM re-reads them hot.
// ============================================================================
__global__ __launch_bounds__(256)
void qtl_prep(const float* __restrict__ x, const int* __restrict__ qw,
              const float* __restrict__ swp,
              _Float16* __restrict__ xh, _Float16* __restrict__ wh,
              long nx4, long nw4)
{
    const float sw = swp[0];
    long tid    = (long)blockIdx.x * blockDim.x + threadIdx.x;
    long stride = (long)gridDim.x * blockDim.x;
    for (long i = tid; i < nx4; i += stride) {
        float4 v = ((const float4*)x)[i];
        h4 hv = { (_Float16)v.x, (_Float16)v.y, (_Float16)v.z, (_Float16)v.w };
        ((h4*)xh)[i] = hv;
    }
    for (long i = tid; i < nw4; i += stride) {
        int4 q = ((const int4*)qw)[i];
        h4 hv = { (_Float16)(q.x * sw), (_Float16)(q.y * sw),
                  (_Float16)(q.z * sw), (_Float16)(q.w * sw) };
        ((h4*)wh)[i] = hv;
    }
}

// ============================================================================
// TDM tile load: tileRows x 32 f16 tile from a [rows, strideElems] f16 tensor
// into LDS at ldsOff. D# per CDNA5 ISA:
//   group0: count=1 | lds_addr | global_addr[56:0] | type=2
//   group1: data_size=1(2B), pad_enable, pad_interval=3(16 dw=64B=one row),
//           pad_amount=3(4 dw=16B) -> effective LDS row stride 80B == KPAD*2.
// ============================================================================
#if QTL_TDM
__device__ __forceinline__
void tdm_load_tile(const _Float16* gsrc, unsigned ldsOff, int strideElems,
                   int rows, int tileRows)
{
    uint64_t ga   = (uint64_t)(uintptr_t)gsrc;
    u32x4 g0 = { 1u,                                   // count=1 (valid)
                 ldsOff,                               // lds_addr (bytes)
                 (unsigned)ga,                         // global_addr[31:0]
                 (unsigned)((ga >> 32) & 0x01FFFFFFull) | 0x80000000u }; // [56:32] | type=2
    unsigned d0   = (1u << 16)    // data_size = 2 bytes
                  | (1u << 20)    // pad_enable
                  | (3u << 22)    // pad_interval: 16 dwords
                  | (3u << 25);   // pad_amount:   4 dwords
    unsigned dim0 = (unsigned)strideElems;
    unsigned dim1 = (unsigned)rows;
    i32x8 g1 = { (int)d0,
                 (int)((dim0 & 0xFFFFu) << 16),                    // dim0[15:0]
                 (int)((dim0 >> 16) | ((dim1 & 0xFFFFu) << 16)),   // dim0 hi | dim1 lo
                 (int)((dim1 >> 16) | ((unsigned)BK << 16)),       // dim1 hi | tile_dim0=32
                 (int)(unsigned)tileRows,                          // tile_dim1, tile_dim2=0
                 (int)(unsigned)strideElems,                       // dim0_stride[31:0]
                 0, 0 };                                           // stride hi / dim1_stride
    i32x4 z4 = { 0, 0, 0, 0 };
#if QTL_TDM_6ARG
    i32x8 z8 = { 0, 0, 0, 0, 0, 0, 0, 0 };
    __builtin_amdgcn_tensor_load_to_lds(g0, g1, z4, z4, z8, 0);
#else
    __builtin_amdgcn_tensor_load_to_lds(g0, g1, z4, z4, 0);
#endif
}
#endif

// ============================================================================
// Kernel 2: f16 WMMA GEMM. y = Xf16 * Wf16^T + bias.
// 128x256 workgroup tile, 8 waves as 2(M)x4(N); each wave 64x64:
// 4x4 16x16 f32 accumulators, 16 WMMA per k-step from 16 ds_load_b128 (1:1),
// 4x register reuse of each fragment. TDM DMA double-buffered staging.
// ============================================================================
__global__ __launch_bounds__(THREADS)
void qtl_gemm_f16(const _Float16* __restrict__ xh, const _Float16* __restrict__ wh,
                  const float* __restrict__ bias, float* __restrict__ out,
                  int M, int N, int K)
{
    __shared__ _Float16 As[2][BM][KPAD];   // 20 KB
    __shared__ _Float16 Bs[2][BN][KPAD];   // 40 KB

    const int t    = threadIdx.x;
    const int lane = t & 31;
    const int wave = t >> 5;
    const int wr   = wave & 1;            // wave row (M), 2 waves
    const int wc   = wave >> 1;           // wave col (N), 4 waves
    const int m0   = blockIdx.y * BM;
    const int n0   = blockIdx.x * BN;

    const _Float16* aBase = xh + (size_t)m0 * K;
    const _Float16* bBase = wh + (size_t)n0 * K;

    v8f acc[4][4] = {};
    const int mW    = wr * 64;
    const int nW    = wc * 64;
    const int lhalf = lane >> 4;          // 0: lanes 0-15, 1: lanes 16-31
    const int l16   = lane & 15;
    const int nk    = K / BK;

#if QTL_TDM
    const unsigned ldsA[2] = { (unsigned)(uintptr_t)&As[0][0][0],
                               (unsigned)(uintptr_t)&As[1][0][0] };
    const unsigned ldsB[2] = { (unsigned)(uintptr_t)&Bs[0][0][0],
                               (unsigned)(uintptr_t)&Bs[1][0][0] };
    if (wave == 0) tdm_load_tile(aBase, ldsA[0], K, M - m0, BM);
    if (wave == 1) tdm_load_tile(bBase, ldsB[0], K, N - n0, BN);
    __builtin_amdgcn_s_wait_tensorcnt(0);
#else
    // non-TDM staging: rows x 4 h8-chunks, distributed over 256 threads
    auto stage_fb = [&](const _Float16* src, _Float16 (*dst)[KPAD], int chunks) {
        for (int idx = t; idx < chunks; idx += THREADS) {
            int row = idx >> 2;
            int col = (idx & 3) << 3;
            *(h8*)&dst[row][col] = *(const h8*)(src + (size_t)row * K + col);
        }
    };
    stage_fb(aBase, As[0], BM * 4);
    stage_fb(bBase, Bs[0], BN * 4);
#endif
    __syncthreads();

    int buf = 0;
    for (int kt = 0; kt < nk; ++kt) {
        if (kt + 1 < nk) {
#if QTL_TDM
            if (wave == 0) tdm_load_tile(aBase + (kt + 1) * BK, ldsA[buf ^ 1], K, M - m0, BM);
            if (wave == 1) tdm_load_tile(bBase + (kt + 1) * BK, ldsB[buf ^ 1], K, N - n0, BN);
#else
            stage_fb(aBase + (kt + 1) * BK, As[buf ^ 1], BM * 4);
            stage_fb(bBase + (kt + 1) * BK, Bs[buf ^ 1], BN * 4);
#endif
        }

        // A frags: 16x32 f16, lane<16 -> K {0..7,16..23}, lane>=16 -> K {8..15,24..31}
        v16h a[4], b[4];
#pragma unroll
        for (int i = 0; i < 4; ++i) {
            const _Float16* r = &As[buf][mW + i * 16 + l16][lhalf * 8];
            h8 lo = *(const h8*)(r);
            h8 hi = *(const h8*)(r + 16);
            a[i] = __builtin_shufflevector(lo, hi, 0,1,2,3,4,5,6,7,8,9,10,11,12,13,14,15);
        }
        // B frags: 32x16 f16, lane<16 -> K {0..7,16..23} of column n, lane>=16 -> K {8..15,24..31}
#pragma unroll
        for (int j = 0; j < 4; ++j) {
            const _Float16* r = &Bs[buf][nW + j * 16 + l16][lhalf * 8];
            h8 lo = *(const h8*)(r);
            h8 hi = *(const h8*)(r + 16);
            b[j] = __builtin_shufflevector(lo, hi, 0,1,2,3,4,5,6,7,8,9,10,11,12,13,14,15);
        }

#pragma unroll
        for (int i = 0; i < 4; ++i)
#pragma unroll
            for (int j = 0; j < 4; ++j)
                acc[i][j] = __builtin_amdgcn_wmma_f32_16x16x32_f16(
                    false, a[i], false, b[j], (short)0, acc[i][j], false, false);

#if QTL_TDM
        __builtin_amdgcn_s_wait_tensorcnt(0);
#endif
        __syncthreads();
        buf ^= 1;
    }

    // Epilogue: C layout — VGPR r: lanes0-15 M=r, lanes16-31 M=r+8; N=lane%16.
#pragma unroll
    for (int j = 0; j < 4; ++j) {
        const int col  = n0 + nW + j * 16 + l16;
        const float bj = bias[col];
#pragma unroll
        for (int i = 0; i < 4; ++i) {
            const int rbase = m0 + mW + i * 16 + lhalf * 8;
#pragma unroll
            for (int r = 0; r < 8; ++r) {
                __builtin_nontemporal_store(acc[i][j][r] + bj,
                                            out + (size_t)(rbase + r) * N + col);
            }
        }
    }
}

// ============================================================================
// Fallback fused kernel (used if ws can't hold the f16 operands or M % BM / N % BN != 0):
// round-1 design (128x128 tile, 8 waves as 4x2), dequantizing during staging.
// ============================================================================
#define FBM 128
#define FBN 128

__global__ __launch_bounds__(THREADS)
void qtl_gemm_fused(const float* __restrict__ x,  const int* __restrict__ qw,
                    const float* __restrict__ swp, const float* __restrict__ bias,
                    float* __restrict__ out, int M, int N, int K)
{
    __shared__ _Float16 As[2][FBM][KPAD];
    __shared__ _Float16 Bs[2][FBN][KPAD];

    const int t     = threadIdx.x;
    const int lane  = t & 31;
    const int wave  = t >> 5;
    const int wr    = wave & 3;
    const int wc    = wave >> 2;
    const int m0    = blockIdx.y * FBM;
    const int n0    = blockIdx.x * FBN;
    const float sw  = swp[0];

    int ldRow[4], ldCol[4];
#pragma unroll
    for (int v = 0; v < 4; ++v) {
        int idx  = t + v * THREADS;
        ldRow[v] = idx >> 3;
        ldCol[v] = (idx & 7) << 2;
    }

    float4 aR[4];
    int4   bR[4];

    auto load_tile = [&](int kt) {
        const int kb = kt * BK;
#pragma unroll
        for (int v = 0; v < 4; ++v) {
            aR[v] = *(const float4*)(x  + (size_t)(m0 + ldRow[v]) * K + kb + ldCol[v]);
            bR[v] = *(const int4*)  (qw + (size_t)(n0 + ldRow[v]) * K + kb + ldCol[v]);
        }
    };
    auto stage_tile = [&](int buf) {
#pragma unroll
        for (int v = 0; v < 4; ++v) {
            h4 ah = { (_Float16)aR[v].x, (_Float16)aR[v].y,
                      (_Float16)aR[v].z, (_Float16)aR[v].w };
            h4 bh = { (_Float16)((float)bR[v].x * sw), (_Float16)((float)bR[v].y * sw),
                      (_Float16)((float)bR[v].z * sw), (_Float16)((float)bR[v].w * sw) };
            *(h4*)&As[buf][ldRow[v]][ldCol[v]] = ah;
            *(h4*)&Bs[buf][ldRow[v]][ldCol[v]] = bh;
        }
    };

    v8f acc[2][4] = {};
    const int mW    = wr * 32;
    const int nW    = wc * 64;
    const int lhalf = lane >> 4;
    const int l16   = lane & 15;
    const int nk    = K / BK;

    load_tile(0);
    stage_tile(0);
    __syncthreads();

    int buf = 0;
    for (int kt = 0; kt < nk; ++kt) {
        if (kt + 1 < nk) load_tile(kt + 1);

        v16h a[2], b[4];
#pragma unroll
        for (int i = 0; i < 2; ++i) {
            const _Float16* r = &As[buf][mW + i * 16 + l16][lhalf * 8];
            h8 lo = *(const h8*)(r);
            h8 hi = *(const h8*)(r + 16);
            a[i] = __builtin_shufflevector(lo, hi, 0,1,2,3,4,5,6,7,8,9,10,11,12,13,14,15);
        }
#pragma unroll
        for (int j = 0; j < 4; ++j) {
            const _Float16* r = &Bs[buf][nW + j * 16 + l16][lhalf * 8];
            h8 lo = *(const h8*)(r);
            h8 hi = *(const h8*)(r + 16);
            b[j] = __builtin_shufflevector(lo, hi, 0,1,2,3,4,5,6,7,8,9,10,11,12,13,14,15);
        }
#pragma unroll
        for (int i = 0; i < 2; ++i)
#pragma unroll
            for (int j = 0; j < 4; ++j)
                acc[i][j] = __builtin_amdgcn_wmma_f32_16x16x32_f16(
                    false, a[i], false, b[j], (short)0, acc[i][j], false, false);

        if (kt + 1 < nk) stage_tile(buf ^ 1);
        __syncthreads();
        buf ^= 1;
    }

#pragma unroll
    for (int j = 0; j < 4; ++j) {
        const int col  = n0 + nW + j * 16 + l16;
        const float bj = bias[col];
#pragma unroll
        for (int i = 0; i < 2; ++i) {
            const int rbase = m0 + mW + i * 16 + lhalf * 8;
#pragma unroll
            for (int r = 0; r < 8; ++r) {
                __builtin_nontemporal_store(acc[i][j][r] + bj,
                                            out + (size_t)(rbase + r) * N + col);
            }
        }
    }
}

// ============================================================================
extern "C" void kernel_launch(void* const* d_in, const int* in_sizes, int n_in,
                              void* d_out, int out_size, void* d_ws, size_t ws_size,
                              hipStream_t stream) {
    const float* x    = (const float*)d_in[0];   // [B,S,K] fp32
    const int*   qw   = (const int*)  d_in[1];   // [N,K] int8-in-int32
    const float* sw   = (const float*)d_in[2];   // scalar
    const float* bias = (const float*)d_in[3];   // [N]
    float*       out  = (float*)d_out;           // [B,S,N] fp32

    const int N = in_sizes[3];
    const int K = in_sizes[1] / N;
    const int M = in_sizes[0] / K;               // B*S

    const size_t xB = (size_t)M * K * sizeof(_Float16);
    const size_t wB = (size_t)N * K * sizeof(_Float16);

    if (ws_size >= xB + wB && (N % BN) == 0 && (M % BM) == 0) {
        _Float16* xh = (_Float16*)d_ws;
        _Float16* wh = (_Float16*)((char*)d_ws + xB);
        long nx4 = (long)M * K / 4;
        long nw4 = (long)N * K / 4;
        qtl_prep<<<4096, 256, 0, stream>>>(x, qw, sw, xh, wh, nx4, nw4);
        dim3 grid(N / BN, M / BM);               // 16 x 64 workgroups
        qtl_gemm_f16<<<grid, THREADS, 0, stream>>>(xh, wh, bias, out, M, N, K);
    } else {
        dim3 grid_f(N / FBN, M / FBM);
        qtl_gemm_fused<<<grid_f, THREADS, 0, stream>>>(x, qw, sw, bias, out, M, N, K);
    }
}
